// RecurrentGemma_87308095193563
// MI455X (gfx1250) — matrix operand order repaired and checked by reading, not compiled, not run on hardware
//
#include <hip/hip_runtime.h>
#include <cstdint>
#include <cstddef>

// ---------------- model constants ----------------
#define S_LEN    2048
#define BATCH    2
#define TOKENS   (S_LEN * BATCH)   // 4096 rows
#define WIDTH    1024
#define EXPANDED 3072
#define NHEADS   8
#define HDIM     128
#define LRU_W    1024
#define BLKD     128               // LRU / NUM_HEADS
#define WINDOW   1024
#define CAPV     30.0f

// ---------------- WMMA types ----------------
typedef __attribute__((ext_vector_type(16))) _Float16 v16h;
typedef __attribute__((ext_vector_type(8)))  _Float16 v8h;
typedef __attribute__((ext_vector_type(2)))  __fp16   pk2;   // matches cvt_pkrtz return
typedef __attribute__((ext_vector_type(8)))  float    v8f;

// ---------------- device helpers ----------------
__device__ __forceinline__ float gelu_f(float x) {
  float x3 = x * x * x;
  return 0.5f * x * (1.0f + tanhf(0.7978845608028654f * (x + 0.044715f * x3)));
}
__device__ __forceinline__ float sigmoid_f(float x) {
  return 1.0f / (1.0f + __expf(-x));
}

// ================= WMMA GEMM =================
// C[M,N] = epi( A[M,K] (f32, lda) * B (f32, ldb; BT? [N,K] : [K,N]) + bias )
// 64x128 output tile / block, BK=32, 256 threads = 8 waves.
// Wave (mt, nhalf) owns one A fragment (16x32) x 4 B fragments -> 4 WMMAs/K-step.
// LDS double-buffered; A stored [m][k], B stored TRANSPOSED [n][k] so every
// fragment is two contiguous 16-byte ds_load_b128 reads.
#define BM 64
#define BN 128
#define BK 32
#define BKP 48   // padded LDS row stride in f16 (96 bytes: 16B-aligned, bank-spread)

// EPI: 0=bias only, 1=bias+gelu, 2=bias+sigmoid, 3=cap*tanh(x/cap)
template<int EPI, bool BT>
__global__ __launch_bounds__(256)
void gemm_wmma(const float* __restrict__ A, int lda,
               const float* __restrict__ B, int ldb,
               const float* __restrict__ bias,
               float* __restrict__ C, int ldc,
               int M, int N, int K) {
  __shared__ _Float16 As[2][BM * BKP];
  __shared__ _Float16 Bs[2][BN * BKP];

  const int tid  = threadIdx.x;
  const int wv   = tid >> 5;
  const int lane = tid & 31;
  const int hi   = lane >> 4;        // half-wave selector
  const int l16  = lane & 15;
  const int mt   = wv >> 1;          // 0..3 : m-tile of this wave
  const int nt0  = (wv & 1) * 4;     // 0 or 4: first of 4 n-tiles for this wave
  const int row0 = blockIdx.y * BM;
  const int col0 = blockIdx.x * BN;

  v8f acc[4] = {};
  float2 ra[4];
  float  rb0[8], rb1[8];

  // ---- stage K-slab k0 into registers (global loads) ----
  auto loadRegs = [&](int k0) {
    #pragma unroll
    for (int r = 0; r < 4; ++r) {           // A: 64 rows x 16 k-pairs
      int i  = tid + r * 256;
      int m  = i >> 4;
      int kp = i & 15;
      ra[r] = *(const float2*)&A[(size_t)(row0 + m) * lda + k0 + 2 * kp];
    }
    #pragma unroll
    for (int r = 0; r < 8; ++r) {           // B: 16 k-pairs x 128 cols
      int i  = tid + r * 256;
      int kp = i >> 7;
      int n  = i & 127;
      if (BT) {
        float2 t = *(const float2*)&B[(size_t)(col0 + n) * ldb + k0 + 2 * kp];
        rb0[r] = t.x; rb1[r] = t.y;
      } else {
        rb0[r] = B[(size_t)(k0 + 2 * kp) * ldb + col0 + n];
        rb1[r] = B[(size_t)(k0 + 2 * kp + 1) * ldb + col0 + n];
      }
    }
    // speculative prefetch of the following A slab (global_prefetch_b8)
    __builtin_prefetch(&A[(size_t)(row0 + (tid >> 2)) * lda + k0 + BK], 0, 0);
  };

  // ---- registers -> LDS (packed f16) ----
  auto storeLds = [&](int buf) {
    #pragma unroll
    for (int r = 0; r < 4; ++r) {
      int i  = tid + r * 256;
      int m  = i >> 4;
      int kp = i & 15;
      *(pk2*)&As[buf][m * BKP + 2 * kp] = __builtin_amdgcn_cvt_pkrtz(ra[r].x, ra[r].y);
    }
    #pragma unroll
    for (int r = 0; r < 8; ++r) {
      int i  = tid + r * 256;
      int kp = i >> 7;
      int n  = i & 127;
      *(pk2*)&Bs[buf][n * BKP + 2 * kp] = __builtin_amdgcn_cvt_pkrtz(rb0[r], rb1[r]);
    }
  };

  const int ar = mt * 16 + l16;
  const int nkq = K / BK;

  loadRegs(0);
  for (int kt = 0; kt < nkq; ++kt) {
    const int buf = kt & 1;
    storeLds(buf);
    __syncthreads();
    if (kt + 1 < nkq) loadRegs((kt + 1) * BK);   // overlap with WMMAs below

    // A fragment: two contiguous 16B runs (ISA 16-bit A 16x32 layout)
    const _Float16* Ab = As[buf];
    v8h alo = *(const v8h*)&Ab[ar * BKP + hi * 8];
    v8h ahi = *(const v8h*)&Ab[ar * BKP + 16 + hi * 8];
    v16h af = __builtin_shufflevector(alo, ahi, 0, 1, 2, 3, 4, 5, 6, 7,
                                      8, 9, 10, 11, 12, 13, 14, 15);
    const _Float16* Bb = Bs[buf];
    #pragma unroll
    for (int j = 0; j < 4; ++j) {
      const _Float16* bp = &Bb[(size_t)((nt0 + j) * 16 + l16) * BKP + hi * 8];
      v8h blo = *(const v8h*)bp;
      v8h bhi = *(const v8h*)(bp + 16);
      v16h bf = __builtin_shufflevector(blo, bhi, 0, 1, 2, 3, 4, 5, 6, 7,
                                        8, 9, 10, 11, 12, 13, 14, 15);
      acc[j] = __builtin_amdgcn_wmma_f32_16x16x32_f16(false, af, false, bf,
                                                      (short)0, acc[j], false, false);
    }
    __syncthreads();
  }

  // ---- epilogue (C/D layout: M = r + 8*hi, N = lane&15) ----
  const int orow = row0 + mt * 16 + hi * 8;
  #pragma unroll
  for (int j = 0; j < 4; ++j) {
    int oc = col0 + (nt0 + j) * 16 + l16;
    float bz = bias ? bias[oc] : 0.0f;
    #pragma unroll
    for (int r = 0; r < 8; ++r) {
      float v = acc[j][r] + bz;
      if (EPI == 1)      v = gelu_f(v);
      else if (EPI == 2) v = sigmoid_f(v);
      else if (EPI == 3) v = CAPV * tanhf(v / CAPV);
      C[(size_t)(orow + r) * ldc + oc] = v;
    }
  }
}

// ================= elementwise / misc kernels =================
__global__ void embed_kernel(const int* __restrict__ tokens,
                             const float* __restrict__ emb,
                             float* __restrict__ x) {
  int row = blockIdx.x;
  int tok = tokens[row];
  for (int c = threadIdx.x; c < WIDTH; c += 256)
    x[(size_t)row * WIDTH + c] = emb[(size_t)tok * WIDTH + c] * 32.0f; // sqrt(1024)
}

__global__ void rmsnorm_kernel(const float* __restrict__ x,
                               const float* __restrict__ scale,
                               float* __restrict__ out) {
  __shared__ float red[256];
  int row = blockIdx.x, t = threadIdx.x;
  const float* xr = x + (size_t)row * WIDTH;
  float s = 0.0f;
  for (int c = t; c < WIDTH; c += 256) { float v = xr[c]; s += v * v; }
  red[t] = s; __syncthreads();
  for (int k = 128; k > 0; k >>= 1) { if (t < k) red[t] += red[t + k]; __syncthreads(); }
  float inv = rsqrtf(red[0] / (float)WIDTH + 1e-6f);
  for (int c = t; c < WIDTH; c += 256)
    out[(size_t)row * WIDTH + c] = xr[c] * inv * (1.0f + scale[c]);
}

__global__ void add_kernel(float* __restrict__ dst, const float* __restrict__ a,
                           const float* __restrict__ b, int n) {
  int i = blockIdx.x * 256 + threadIdx.x;
  if (i < n) dst[i] = a[i] + b[i];
}
__global__ void mul_kernel(float* __restrict__ dst, const float* __restrict__ a,
                           const float* __restrict__ b, int n) {
  int i = blockIdx.x * 256 + threadIdx.x;
  if (i < n) dst[i] = a[i] * b[i];
}

__global__ void conv_kernel(const float* __restrict__ in, const float* __restrict__ w,
                            const float* __restrict__ b, float* __restrict__ out) {
  int idx = blockIdx.x * 256 + threadIdx.x;
  if (idx >= TOKENS * LRU_W) return;
  int ch  = idx & (LRU_W - 1);
  int row = idx >> 10;
  int s   = row & (S_LEN - 1);
  float acc = b[ch];
  #pragma unroll
  for (int i = 0; i < 4; ++i) {
    int sp = s - 3 + i;
    if (sp >= 0) acc += in[(size_t)(row - 3 + i) * LRU_W + ch] * w[i * LRU_W + ch];
  }
  out[idx] = acc;
}

// RG-LRU sequential scan: one thread per (batch, channel)
__global__ void scan_kernel(const float* __restrict__ hconv,
                            const float* __restrict__ gx,
                            const float* __restrict__ ga,
                            const float* __restrict__ a_param,
                            float* __restrict__ out) {
  int tid = blockIdx.x * 256 + threadIdx.x;
  if (tid >= BATCH * LRU_W) return;
  int b  = tid >> 10;
  int ch = tid & (LRU_W - 1);
  float ap = a_param[ch];
  float sp = (ap > 20.0f) ? ap : log1pf(__expf(ap));   // softplus
  float st = 0.0f;
  for (int s = 0; s < S_LEN; ++s) {
    size_t o  = ((size_t)(b * S_LEN + s)) * LRU_W + ch;
    float la  = -8.0f * ga[o] * sp;
    float a   = __expf(la);
    float mlt = sqrtf(fmaxf(0.0f, 1.0f - __expf(2.0f * la)));
    if (s == 0) { a = 0.0f; mlt = 1.0f; }
    float bb = hconv[o] * gx[o] * mlt;
    st = a * st + bb;
    out[o] = st;
  }
}

// RoPE in-place on (TOKENS, heads, 128); rotates first 64 dims
__global__ void rope_kernel(float* __restrict__ x, int heads) {
  int idx = blockIdx.x * 256 + threadIdx.x;
  int total = TOKENS * heads * 32;
  if (idx >= total) return;
  int i   = idx & 31;
  int rh  = idx >> 5;          // row*heads + h
  int row = rh / heads;
  int s   = row & (S_LEN - 1);
  float fe = 2.0f * (float)i / 64.0f;
  float ts = powf(10000.0f, fe);
  float tt = (float)s / ts;
  float sn = sinf(tt), cs = cosf(tt);
  float* base = x + (size_t)rh * HDIM;
  float a = base[i], b = base[32 + i];
  base[i]      = a * cs - b * sn;
  base[32 + i] = b * cs + a * sn;
}

// Windowed causal MQA attention: one block (128 threads) per (b,h,query)
__global__ __launch_bounds__(128)
void attn_kernel(const float* __restrict__ q, const float* __restrict__ k,
                 const float* __restrict__ v, float* __restrict__ out) {
  __shared__ float qv[HDIM];
  __shared__ float w[WINDOW];
  __shared__ float red[128];
  int row = blockIdx.x;          // 0..TOKENS-1
  int h   = blockIdx.y;
  int b   = row / S_LEN;
  int qs  = row & (S_LEN - 1);
  int t   = threadIdx.x;
  const float scale = 0.08838834764831845f;  // 1/sqrt(128)

  qv[t] = q[((size_t)row * NHEADS + h) * HDIM + t] * scale;
  __syncthreads();

  int j0 = qs - (WINDOW - 1); if (j0 < 0) j0 = 0;
  int nk = qs - j0 + 1;

  for (int jj = t; jj < nk; jj += 128) {
    const float* kr = k + (size_t)(b * S_LEN + j0 + jj) * HDIM;
    float d = 0.0f;
    #pragma unroll 8
    for (int e = 0; e < HDIM; ++e) d += qv[e] * kr[e];
    w[jj] = d;
  }
  __syncthreads();

  float lm = -1e30f;
  for (int jj = t; jj < nk; jj += 128) lm = fmaxf(lm, w[jj]);
  red[t] = lm; __syncthreads();
  for (int s2 = 64; s2 > 0; s2 >>= 1) { if (t < s2) red[t] = fmaxf(red[t], red[t + s2]); __syncthreads(); }
  float mx = red[0]; __syncthreads();

  float ls = 0.0f;
  for (int jj = t; jj < nk; jj += 128) { float e = __expf(w[jj] - mx); w[jj] = e; ls += e; }
  red[t] = ls; __syncthreads();
  for (int s2 = 64; s2 > 0; s2 >>= 1) { if (t < s2) red[t] += red[t + s2]; __syncthreads(); }
  float inv = 1.0f / red[0];
  __syncthreads();

  float acc = 0.0f;
  for (int jj = 0; jj < nk; ++jj)
    acc += w[jj] * v[(size_t)(b * S_LEN + j0 + jj) * HDIM + t];
  out[((size_t)row * NHEADS + h) * HDIM + t] = acc * inv;
}

// ================= host side =================
struct BlkP {
  bool attn;
  const float *pre_scale, *mlp_scale, *w_gate, *b_gate, *w_up, *b_up, *w_down, *b_down;
  const float *w_y, *b_y, *w_x, *b_x, *conv_w, *conv_b, *wi, *bi, *wa, *ba, *a_param, *w_out, *b_out;
  const float *wq, *wk, *wv, *wo, *bo;
};

template<int EPI, bool BT>
static void gemm(hipStream_t st, const float* A, int lda, const float* B, int ldb,
                 const float* bias, float* C, int ldc, int M, int N, int K) {
  dim3 g(N / BN, M / BM), blk(256);
  gemm_wmma<EPI, BT><<<g, blk, 0, st>>>(A, lda, B, ldb, bias, C, ldc, M, N, K);
}

static void run_mlp(hipStream_t st, float* X, float* H, float* T1, float* T2, const BlkP& p) {
  const int n1 = TOKENS * WIDTH;
  rmsnorm_kernel<<<TOKENS, 256, 0, st>>>(X, p.mlp_scale, H);
  gemm<1, false>(st, H, WIDTH, p.w_gate, EXPANDED, p.b_gate, T1, EXPANDED, TOKENS, EXPANDED, WIDTH);
  gemm<0, false>(st, H, WIDTH, p.w_up,   EXPANDED, p.b_up,   T2, EXPANDED, TOKENS, EXPANDED, WIDTH);
  mul_kernel<<<(TOKENS * EXPANDED + 255) / 256, 256, 0, st>>>(T1, T1, T2, TOKENS * EXPANDED);
  gemm<0, false>(st, T1, EXPANDED, p.w_down, WIDTH, p.b_down, H, WIDTH, TOKENS, WIDTH, EXPANDED);
  add_kernel<<<(n1 + 255) / 256, 256, 0, st>>>(X, X, H, n1);
}

extern "C" void kernel_launch(void* const* d_in, const int* in_sizes, int n_in,
                              void* d_out, int out_size, void* d_ws, size_t ws_size,
                              hipStream_t stream) {
  (void)in_sizes; (void)n_in; (void)out_size; (void)ws_size;

  // ---- walk the flattened input dict ----
  int idx = 0;
  const int*   tokens      = (const int*)d_in[idx++];
  const float* emb         = (const float*)d_in[idx++];
  const float* final_scale = (const float*)d_in[idx++];
  BlkP blk[9];
  for (int i = 0; i < 9; ++i) {
    BlkP& p = blk[i];
    p.attn = ((i % 3) == 2);
    p.pre_scale = (const float*)d_in[idx++];
    p.mlp_scale = (const float*)d_in[idx++];
    p.w_gate    = (const float*)d_in[idx++];
    p.b_gate    = (const float*)d_in[idx++];
    p.w_up      = (const float*)d_in[idx++];
    p.b_up      = (const float*)d_in[idx++];
    p.w_down    = (const float*)d_in[idx++];
    p.b_down    = (const float*)d_in[idx++];
    if (!p.attn) {
      p.w_y = (const float*)d_in[idx++];  p.b_y = (const float*)d_in[idx++];
      p.w_x = (const float*)d_in[idx++];  p.b_x = (const float*)d_in[idx++];
      p.conv_w = (const float*)d_in[idx++]; p.conv_b = (const float*)d_in[idx++];
      p.wi = (const float*)d_in[idx++];   p.bi = (const float*)d_in[idx++];
      p.wa = (const float*)d_in[idx++];   p.ba = (const float*)d_in[idx++];
      p.a_param = (const float*)d_in[idx++];
      p.w_out = (const float*)d_in[idx++]; p.b_out = (const float*)d_in[idx++];
    } else {
      p.wq = (const float*)d_in[idx++]; p.wk = (const float*)d_in[idx++];
      p.wv = (const float*)d_in[idx++]; p.wo = (const float*)d_in[idx++];
      p.bo = (const float*)d_in[idx++];
    }
  }

  // ---- workspace layout (floats) ----
  float* W  = (float*)d_ws;
  const size_t NW = (size_t)TOKENS * WIDTH;      // 4M
  const size_t NE = (size_t)TOKENS * EXPANDED;   // 12M
  float* X  = W;
  float* H  = X  + NW;
  float* T1 = H  + NW;     // also gx (first NW) / ga (second NW)
  float* T2 = T1 + NE;
  float* T3 = T2 + NE;
  float* T4 = T3 + NW;
  float* T5 = T4 + NW;

  const int n1 = TOKENS * WIDTH;

  // ---- embedding ----
  embed_kernel<<<TOKENS, 256, 0, stream>>>(tokens, emb, X);

  // ---- blocks ----
  for (int i = 0; i < 9; ++i) {
    const BlkP& p = blk[i];
    rmsnorm_kernel<<<TOKENS, 256, 0, stream>>>(X, p.pre_scale, H);
    if (!p.attn) {
      // y = gelu(H @ w_y + b_y)
      gemm<1, false>(stream, H, WIDTH, p.w_y, LRU_W, p.b_y, T3, LRU_W, TOKENS, LRU_W, WIDTH);
      // h = H @ w_x + b_x ; causal conv
      gemm<0, false>(stream, H, WIDTH, p.w_x, LRU_W, p.b_x, T4, LRU_W, TOKENS, LRU_W, WIDTH);
      conv_kernel<<<(TOKENS * LRU_W + 255) / 256, 256, 0, stream>>>(T4, p.conv_w, p.conv_b, T5);
      // block-diagonal gates via 8 per-head WMMA GEMMs, sigmoid epilogue
      float* gx = T1;
      float* ga = T1 + NW;
      for (int h = 0; h < NHEADS; ++h) {
        gemm<2, false>(stream, T5 + h * BLKD, LRU_W, p.wi + (size_t)h * BLKD * BLKD, BLKD,
                       p.bi + h * BLKD, gx + h * BLKD, LRU_W, TOKENS, BLKD, BLKD);
        gemm<2, false>(stream, T5 + h * BLKD, LRU_W, p.wa + (size_t)h * BLKD * BLKD, BLKD,
                       p.ba + h * BLKD, ga + h * BLKD, LRU_W, TOKENS, BLKD, BLKD);
      }
      scan_kernel<<<(BATCH * LRU_W + 255) / 256, 256, 0, stream>>>(T5, gx, ga, p.a_param, T4);
      mul_kernel<<<(n1 + 255) / 256, 256, 0, stream>>>(T3, T3, T4, n1);
      gemm<0, false>(stream, T3, LRU_W, p.w_out, WIDTH, p.b_out, H, WIDTH, TOKENS, WIDTH, LRU_W);
    } else {
      float* q = T3;
      float* k = T4;
      float* v = T4 + (size_t)TOKENS * HDIM;
      gemm<0, false>(stream, H, WIDTH, p.wq, WIDTH, nullptr, q, WIDTH, TOKENS, WIDTH, WIDTH);
      gemm<0, false>(stream, H, WIDTH, p.wk, HDIM, nullptr, k, HDIM, TOKENS, HDIM, WIDTH);
      gemm<0, false>(stream, H, WIDTH, p.wv, HDIM, nullptr, v, HDIM, TOKENS, HDIM, WIDTH);
      rope_kernel<<<(TOKENS * NHEADS * 32 + 255) / 256, 256, 0, stream>>>(q, NHEADS);
      rope_kernel<<<(TOKENS * 32 + 255) / 256, 256, 0, stream>>>(k, 1);
      dim3 ag(TOKENS, NHEADS);
      attn_kernel<<<ag, 128, 0, stream>>>(q, k, v, T5);
      gemm<0, false>(stream, T5, WIDTH, p.wo, WIDTH, p.bo, H, WIDTH, TOKENS, WIDTH, WIDTH);
    }
    add_kernel<<<(n1 + 255) / 256, 256, 0, stream>>>(X, X, H, n1);
    run_mlp(stream, X, H, T1, T2, p);
  }

  // ---- final norm + capped logits (B^T GEMM against embedding) ----
  rmsnorm_kernel<<<TOKENS, 256, 0, stream>>>(X, final_scale, H);
  gemm<3, true>(stream, H, WIDTH, emb, WIDTH, nullptr, (float*)d_out, 256, TOKENS, 256, WIDTH);
}
